// CausalSemigroupSelfAttentionSelective_39651138076898
// MI455X (gfx1250) — matrix operand order repaired and checked by reading, not compiled, not run on hardware
//
#include <hip/hip_runtime.h>
#include <math.h>

// ---------------------------------------------------------------------------
// CDNA5 (gfx1250) causal attention with diffusion prior + truncated Poisson
// semigroup, built on v_wmma_f32_16x16x32_bf16 (wave32).
//
//   K1 qkv_rope   : x @ W_qkv (bf16 WMMA), RoPE on q,k, scatter to [H,T,64]
//   K2 attn pass1 : S^T = K Q^T flash softmax (per-lane stats), BN=64 keys,
//                   register-double-buffered staging, pv1 = P @ v (f32)
//   K3 attn pass2 : rebuild P from stats, pv2 = P @ pv1, fuse
//                   y = w0*v + w1*pv1 + w2*pv2 -> ybuf [T,768] bf16
//   K4 proj       : ybuf @ W_proj -> f32 out
// ---------------------------------------------------------------------------

typedef __bf16 bf16_t;
typedef __attribute__((ext_vector_type(16))) __bf16 v16bf;
typedef __attribute__((ext_vector_type(8)))  __bf16 v8bf;
typedef __attribute__((ext_vector_type(4)))  __bf16 v4bf;
typedef __attribute__((ext_vector_type(8)))  float  v8f;

#define HN 12
#define TT 4096
#define DH 64
#define CC 768

__device__ __forceinline__ bf16_t f2bf(float f) {
    union { float f; unsigned u; } a; a.f = f;
    unsigned u = a.u;
    unsigned r = (u + 0x7FFFu + ((u >> 16) & 1u)) >> 16;   // RNE
    union { unsigned short s; bf16_t b; } o; o.s = (unsigned short)r;
    return o.b;
}
__device__ __forceinline__ float bf2f(bf16_t b) {
    union { unsigned short s; bf16_t b; } i; i.b = b;
    union { float f; unsigned u; } o; o.u = ((unsigned)i.s) << 16;
    return o.f;
}
__device__ __forceinline__ v8f zero8() {
    v8f z;
#pragma unroll
    for (int i = 0; i < 8; ++i) z[i] = 0.0f;
    return z;
}
__device__ __forceinline__ v16bf make16(v8bf lo, v8bf hi) {
    return __builtin_shufflevector(lo, hi, 0,1,2,3,4,5,6,7,8,9,10,11,12,13,14,15);
}
__device__ __forceinline__ v8f wmma_bf16(v16bf a, v16bf b, v8f c) {
    return __builtin_amdgcn_wmma_f32_16x16x32_bf16(false, a, false, b, (short)0, c, false, false);
}

// A-fragment (16 rows x 32 K), source row-major [row][k], stride in elements.
__device__ __forceinline__ v16bf load_a_frag(const bf16_t* base, int stride, int lane) {
    int m = lane & 15, grp = lane >> 4;
    const bf16_t* row = base + m * stride;
    v8bf lo = *(const v8bf*)(row + grp * 8);
    v8bf hi = *(const v8bf*)(row + 16 + grp * 8);
    return make16(lo, hi);
}
// B-fragment (32 K x 16 n), source stored transposed Bt[n][k], stride in elems.
__device__ __forceinline__ v16bf load_b_frag(const bf16_t* baseT, int stride, int lane) {
    int n = lane & 15, grp = lane >> 4;
    const bf16_t* row = baseT + n * stride;
    v8bf lo = *(const v8bf*)(row + grp * 8);
    v8bf hi = *(const v8bf*)(row + 16 + grp * 8);
    return make16(lo, hi);
}

// ---------------------------------------------------------------------------
// K1: qkv = x @ W_qkv, RoPE(q,k), scatter to [H,T,DH] bf16.
// grid (T/128, 36)
// ---------------------------------------------------------------------------
__global__ __launch_bounds__(256) void qkv_rope_kernel(
    const float* __restrict__ x, const float* __restrict__ Wqkv,
    const float* __restrict__ cosT, const float* __restrict__ sinT,
    bf16_t* __restrict__ qh, bf16_t* __restrict__ kh, bf16_t* __restrict__ vh)
{
    __shared__ bf16_t As[128][32];
    __shared__ bf16_t Bt[64][40];
    int tid  = threadIdx.x;
    int lane = tid & 31, wave = tid >> 5;
    int grp  = lane >> 4, nl = lane & 15;
    int m0   = blockIdx.x * 128;
    int g    = blockIdx.y;
    int n0   = g * 64;

    v8f acc[4] = {};

    for (int k0 = 0; k0 < CC; k0 += 32) {
        // prefetch next tiles (uniform cond, hoisted, near-cache locality)
        if (k0 + 32 < CC) {
            __builtin_prefetch(&x[(size_t)(m0 + (tid >> 1)) * CC + k0 + 32 + (tid & 1) * 16], 0, 3);
            __builtin_prefetch(&Wqkv[(size_t)(k0 + 32 + (tid >> 3)) * (3 * CC) + n0 + (tid & 7) * 8], 0, 3);
        }
        // stage A (x tile, f32 -> bf16): 4 constant-trip iterations
#pragma unroll
        for (int it = 0; it < 4; ++it) {
            int i = it * 1024 + tid * 4;
            int r = i >> 5, c = i & 31;
            float4 f = *(const float4*)&x[(size_t)(m0 + r) * CC + k0 + c];
            v4bf b4; b4[0] = f2bf(f.x); b4[1] = f2bf(f.y);
                     b4[2] = f2bf(f.z); b4[3] = f2bf(f.w);
            *(v4bf*)&As[r][c] = b4;
        }
        // stage B transposed: 2 constant-trip iterations
#pragma unroll
        for (int it = 0; it < 2; ++it) {
            int i = it * 1024 + tid * 4;
            int kk = i >> 6, nn = i & 63;
            float4 f = *(const float4*)&Wqkv[(size_t)(k0 + kk) * (3 * CC) + n0 + nn];
            Bt[nn + 0][kk] = f2bf(f.x); Bt[nn + 1][kk] = f2bf(f.y);
            Bt[nn + 2][kk] = f2bf(f.z); Bt[nn + 3][kk] = f2bf(f.w);
        }
        __syncthreads();
        v16bf a = load_a_frag(&As[wave * 16][0], 32, lane);
        v16bf bfr[4];
#pragma unroll
        for (int s = 0; s < 4; ++s) bfr[s] = load_b_frag(&Bt[s * 16][0], 40, lane);
#pragma unroll
        for (int s = 0; s < 4; ++s) acc[s] = wmma_bf16(a, bfr[s], acc[s]);
        __syncthreads();
    }

    int type = g / HN, h = g % HN;
    bf16_t* dst = (type == 0) ? qh : (type == 1) ? kh : vh;
#pragma unroll
    for (int s = 0; s < 4; ++s) {
#pragma unroll
        for (int r = 0; r < 8; ++r) {
            int t = m0 + wave * 16 + grp * 8 + r;
            int d = s * 16 + nl;
            float val = acc[s][r];
            if (type < 2) {  // RoPE: partner column d^32 lives in subtile s^2
                float partner = acc[s ^ 2][r];
                float rot = (d < 32) ? -partner : partner;
                val = val * cosT[t * DH + d] + rot * sinT[t * DH + d];
            }
            dst[(size_t)(h * TT + t) * DH + d] = f2bf(val);
        }
    }
}

// ---------------------------------------------------------------------------
// K2: flash pass 1. S^T = K Q^T (per-lane stats), BN=64 keys per step,
// register double-buffered K/V staging. grid (T/128, H).
// ---------------------------------------------------------------------------
__global__ __launch_bounds__(256) void attn_pass1_kernel(
    const bf16_t* __restrict__ qh, const bf16_t* __restrict__ kh,
    const bf16_t* __restrict__ vh,
    float* __restrict__ pv1, float* __restrict__ ms, float* __restrict__ ls,
    const float* __restrict__ kappa_un, const float* __restrict__ xi_un)
{
    __shared__ bf16_t Kt[64][72];      // [key][dh]
    __shared__ bf16_t Vt[64][72];      // [dh][key]
    __shared__ bf16_t Ps[8][16][64];   // per-wave [query][key]

    int tid  = threadIdx.x;
    int lane = tid & 31, wave = tid >> 5;
    int grp  = lane >> 4, nl = lane & 15;
    int h    = blockIdx.y;
    int m0   = blockIdx.x * 128;

    float kappa = log1pf(expf(kappa_un[0]));
    float xi    = log1pf(expf(xi_un[0]));
    float pc    = kappa / (xi * xi);
    const float scale = 0.125f;  // 1/sqrt(64)

    const bf16_t* qbase = qh + (size_t)(h * TT + m0 + wave * 16) * DH;
    v16bf qb0 = load_b_frag(qbase,      DH, lane);   // dh 0..31
    v16bf qb1 = load_b_frag(qbase + 32, DH, lane);   // dh 32..63

    int tq = m0 + wave * 16 + nl;
    float m_i = -1e30f, l_i = 0.0f;
    v8f o[4];
#pragma unroll
    for (int s = 0; s < 4; ++s) o[s] = zero8();

    int sk = tid >> 2;            // staged key row 0..63
    int sd = (tid & 3) * 16;      // staged dh chunk base
    int kend = m0 + 128;
    int wave_hi = m0 + wave * 16 + 15;

    // preload tile 0 into registers
    v8bf kr0, kr1, vr0, vr1;
    {
        const bf16_t* kp = kh + (size_t)(h * TT + sk) * DH + sd;
        const bf16_t* vp = vh + (size_t)(h * TT + sk) * DH + sd;
        kr0 = *(const v8bf*)kp; kr1 = *(const v8bf*)(kp + 8);
        vr0 = *(const v8bf*)vp; vr1 = *(const v8bf*)(vp + 8);
    }

    for (int kb = 0; kb < kend; kb += 64) {
        // commit staged registers to LDS
        *(v8bf*)&Kt[sk][sd]     = kr0;
        *(v8bf*)&Kt[sk][sd + 8] = kr1;
#pragma unroll
        for (int j = 0; j < 8; ++j) Vt[sd + j][sk]     = vr0[j];
#pragma unroll
        for (int j = 0; j < 8; ++j) Vt[sd + 8 + j][sk] = vr1[j];
        // issue next-tile global loads; they overlap compute below
        if (kb + 64 < kend) {
            const bf16_t* kp = kh + (size_t)(h * TT + kb + 64 + sk) * DH + sd;
            const bf16_t* vp = vh + (size_t)(h * TT + kb + 64 + sk) * DH + sd;
            kr0 = *(const v8bf*)kp; kr1 = *(const v8bf*)(kp + 8);
            vr0 = *(const v8bf*)vp; vr1 = *(const v8bf*)(vp + 8);
        }
        __syncthreads();

        if (kb <= wave_hi) {  // wave-uniform: skip fully-masked key tiles
            // S^T(64 keys x 16 queries): 4 key subtiles x 2 dh halves
            v8f st[4];
#pragma unroll
            for (int sm = 0; sm < 4; ++sm) {
                v16bf ka0 = load_a_frag(&Kt[sm * 16][0],  72, lane);
                v16bf ka1 = load_a_frag(&Kt[sm * 16][32], 72, lane);
                st[sm] = zero8();
                st[sm] = wmma_bf16(ka0, qb0, st[sm]);
                st[sm] = wmma_bf16(ka1, qb1, st[sm]);
            }
            // logits + prior + mask (in place); key = kb + sm*16 + grp*8 + r
            float vmax = -1e30f;
#pragma unroll
            for (int sm = 0; sm < 4; ++sm)
#pragma unroll
                for (int r = 0; r < 8; ++r) {
                    int tk = kb + sm * 16 + grp * 8 + r;
                    float dist = (float)(tq - tk);
                    float v = st[sm][r] * scale - pc * dist * dist;
                    v = (tk <= tq) ? v : -1e30f;
                    st[sm][r] = v;
                    vmax = fmaxf(vmax, v);
                }
            vmax = fmaxf(vmax, __shfl_xor(vmax, 16, 32));
            float mnew = fmaxf(m_i, vmax);
            float pscale_q = __expf(m_i - mnew);
            m_i = mnew;
            float rsum = 0.0f;
#pragma unroll
            for (int sm = 0; sm < 4; ++sm) {
                v8bf pk;
#pragma unroll
                for (int r = 0; r < 8; ++r) {
                    float p = (st[sm][r] > -0.5e30f) ? __expf(st[sm][r] - mnew) : 0.0f;
                    pk[r] = f2bf(p);
                    rsum += p;
                }
                *(v8bf*)&Ps[wave][nl][sm * 16 + grp * 8] = pk;  // ds_store_b128
            }
            rsum += __shfl_xor(rsum, 16, 32);
            l_i = l_i * pscale_q + rsum;

            // broadcast rescale factors to O rows, rescale, O += P @ V
            float oscale[8];
#pragma unroll
            for (int r = 0; r < 8; ++r) oscale[r] = __shfl(pscale_q, grp * 8 + r, 32);
#pragma unroll
            for (int s = 0; s < 4; ++s)
#pragma unroll
                for (int r = 0; r < 8; ++r) o[s][r] *= oscale[r];

            v16bf pa0 = load_a_frag(&Ps[wave][0][0],      64, lane);  // keys 0..31
            v16bf pa1 = load_a_frag(&Ps[wave][0][0] + 32, 64, lane);  // keys 32..63
#pragma unroll
            for (int s = 0; s < 4; ++s) {
                v16bf b0 = load_b_frag(&Vt[s * 16][0],  72, lane);
                v16bf b1 = load_b_frag(&Vt[s * 16][32], 72, lane);
                o[s] = wmma_bf16(pa0, b0, o[s]);
                o[s] = wmma_bf16(pa1, b1, o[s]);
            }
        }
        __syncthreads();
    }

    float linv_q = 1.0f / l_i;
    float linv[8];
#pragma unroll
    for (int r = 0; r < 8; ++r) linv[r] = __shfl(linv_q, grp * 8 + r, 32);
#pragma unroll
    for (int r = 0; r < 8; ++r) {
        int t = m0 + wave * 16 + grp * 8 + r;
#pragma unroll
        for (int s = 0; s < 4; ++s)
            pv1[(size_t)(h * TT + t) * DH + s * 16 + nl] = o[s][r] * linv[r];
    }
    if (lane < 16) { ms[h * TT + tq] = m_i; ls[h * TT + tq] = l_i; }
}

// ---------------------------------------------------------------------------
// K3: pass 2: rebuild normalized P from stats, pv2 = P @ pv1, fuse combine:
//     ybuf[t, h*64+d] = w0*v + w1*pv1 + w2*pv2  (bf16)
// ---------------------------------------------------------------------------
__global__ __launch_bounds__(256) void attn_pass2_kernel(
    const bf16_t* __restrict__ qh, const bf16_t* __restrict__ kh,
    const bf16_t* __restrict__ vh, const float* __restrict__ pv1,
    const float* __restrict__ ms, const float* __restrict__ ls,
    const float* __restrict__ kappa_un, const float* __restrict__ xi_un,
    const float* __restrict__ dt_logit, bf16_t* __restrict__ ybuf)
{
    __shared__ bf16_t Kt[64][72];      // [key][dh]
    __shared__ bf16_t Vt[64][72];      // [dh][key] of pv1 (bf16)
    __shared__ bf16_t Ps[8][16][64];

    int tid  = threadIdx.x;
    int lane = tid & 31, wave = tid >> 5;
    int grp  = lane >> 4, nl = lane & 15;
    int h    = blockIdx.y;
    int m0   = blockIdx.x * 128;

    float kappa = log1pf(expf(kappa_un[0]));
    float xi    = log1pf(expf(xi_un[0]));
    float pc    = kappa / (xi * xi);
    const float scale = 0.125f;

    const bf16_t* qbase = qh + (size_t)(h * TT + m0 + wave * 16) * DH;
    v16bf qb0 = load_b_frag(qbase,      DH, lane);
    v16bf qb1 = load_b_frag(qbase + 32, DH, lane);

    int tq = m0 + wave * 16 + nl;
    float m_i    = ms[h * TT + tq];
    float linv_q = 1.0f / ls[h * TT + tq];
    v8f o[4];
#pragma unroll
    for (int s = 0; s < 4; ++s) o[s] = zero8();

    int sk = tid >> 2, sd = (tid & 3) * 16;
    int kend = m0 + 128;
    int wave_hi = m0 + wave * 16 + 15;

    v8bf kr0, kr1;
    float4 pr0, pr1, pr2, pr3;
    {
        const bf16_t* kp = kh  + (size_t)(h * TT + sk) * DH + sd;
        const float*  pp = pv1 + (size_t)(h * TT + sk) * DH + sd;
        kr0 = *(const v8bf*)kp; kr1 = *(const v8bf*)(kp + 8);
        pr0 = *(const float4*)pp;       pr1 = *(const float4*)(pp + 4);
        pr2 = *(const float4*)(pp + 8); pr3 = *(const float4*)(pp + 12);
    }

    for (int kb = 0; kb < kend; kb += 64) {
        *(v8bf*)&Kt[sk][sd]     = kr0;
        *(v8bf*)&Kt[sk][sd + 8] = kr1;
        Vt[sd +  0][sk] = f2bf(pr0.x); Vt[sd +  1][sk] = f2bf(pr0.y);
        Vt[sd +  2][sk] = f2bf(pr0.z); Vt[sd +  3][sk] = f2bf(pr0.w);
        Vt[sd +  4][sk] = f2bf(pr1.x); Vt[sd +  5][sk] = f2bf(pr1.y);
        Vt[sd +  6][sk] = f2bf(pr1.z); Vt[sd +  7][sk] = f2bf(pr1.w);
        Vt[sd +  8][sk] = f2bf(pr2.x); Vt[sd +  9][sk] = f2bf(pr2.y);
        Vt[sd + 10][sk] = f2bf(pr2.z); Vt[sd + 11][sk] = f2bf(pr2.w);
        Vt[sd + 12][sk] = f2bf(pr3.x); Vt[sd + 13][sk] = f2bf(pr3.y);
        Vt[sd + 14][sk] = f2bf(pr3.z); Vt[sd + 15][sk] = f2bf(pr3.w);
        if (kb + 64 < kend) {
            const bf16_t* kp = kh  + (size_t)(h * TT + kb + 64 + sk) * DH + sd;
            const float*  pp = pv1 + (size_t)(h * TT + kb + 64 + sk) * DH + sd;
            kr0 = *(const v8bf*)kp; kr1 = *(const v8bf*)(kp + 8);
            pr0 = *(const float4*)pp;       pr1 = *(const float4*)(pp + 4);
            pr2 = *(const float4*)(pp + 8); pr3 = *(const float4*)(pp + 12);
        }
        __syncthreads();

        if (kb <= wave_hi) {
            v8f st[4];
#pragma unroll
            for (int sm = 0; sm < 4; ++sm) {
                v16bf ka0 = load_a_frag(&Kt[sm * 16][0],  72, lane);
                v16bf ka1 = load_a_frag(&Kt[sm * 16][32], 72, lane);
                st[sm] = zero8();
                st[sm] = wmma_bf16(ka0, qb0, st[sm]);
                st[sm] = wmma_bf16(ka1, qb1, st[sm]);
            }
            // exact normalized P rows from saved stats; all per-lane math
#pragma unroll
            for (int sm = 0; sm < 4; ++sm) {
                v8bf pk;
#pragma unroll
                for (int r = 0; r < 8; ++r) {
                    int tk = kb + sm * 16 + grp * 8 + r;
                    float dist = (float)(tq - tk);
                    float v = st[sm][r] * scale - pc * dist * dist;
                    float p = (tk <= tq) ? __expf(v - m_i) * linv_q : 0.0f;
                    pk[r] = f2bf(p);
                }
                *(v8bf*)&Ps[wave][nl][sm * 16 + grp * 8] = pk;
            }
            v16bf pa0 = load_a_frag(&Ps[wave][0][0],      64, lane);
            v16bf pa1 = load_a_frag(&Ps[wave][0][0] + 32, 64, lane);
#pragma unroll
            for (int s = 0; s < 4; ++s) {
                v16bf b0 = load_b_frag(&Vt[s * 16][0],  72, lane);
                v16bf b1 = load_b_frag(&Vt[s * 16][32], 72, lane);
                o[s] = wmma_bf16(pa0, b0, o[s]);
                o[s] = wmma_bf16(pa1, b1, o[s]);
            }
        }
        __syncthreads();
    }

    // truncated, renormalized Poisson weights
    float dt = 1.0f / (1.0f + __expf(-dt_logit[0]));
    float e  = __expf(-dt);
    float w0 = e, w1 = dt * e, w2 = 0.5f * dt * dt * e;
    float wsum = w0 + w1 + w2;
    w0 /= wsum; w1 /= wsum; w2 /= wsum;

#pragma unroll
    for (int r = 0; r < 8; ++r) {
        int t = m0 + wave * 16 + grp * 8 + r;
#pragma unroll
        for (int s = 0; s < 4; ++s) {
            int d = s * 16 + nl;
            size_t idx = (size_t)(h * TT + t) * DH + d;
            float y = w0 * bf2f(vh[idx]) + w1 * pv1[idx] + w2 * o[s][r];
            ybuf[(size_t)t * CC + h * DH + d] = f2bf(y);
        }
    }
}

// ---------------------------------------------------------------------------
// K4: out = ybuf @ W_proj  (f32 output). grid (T/128, C/64)
// ---------------------------------------------------------------------------
__global__ __launch_bounds__(256) void proj_kernel(
    const bf16_t* __restrict__ ybuf, const float* __restrict__ Wp,
    float* __restrict__ out)
{
    __shared__ bf16_t As[128][32];
    __shared__ bf16_t Bt[64][40];
    int tid  = threadIdx.x;
    int lane = tid & 31, wave = tid >> 5;
    int grp  = lane >> 4, nl = lane & 15;
    int m0   = blockIdx.x * 128;
    int n0   = blockIdx.y * 64;

    v8f acc[4] = {};

    for (int k0 = 0; k0 < CC; k0 += 32) {
        if (k0 + 32 < CC) {
            __builtin_prefetch(&ybuf[(size_t)(m0 + (tid & 127)) * CC + k0 + 32], 0, 3);
            __builtin_prefetch(&Wp[(size_t)(k0 + 32 + (tid >> 3)) * CC + n0 + (tid & 7) * 8], 0, 3);
        }
#pragma unroll
        for (int it = 0; it < 2; ++it) {
            int i = it * 2048 + tid * 8;
            int r = i >> 5, c = i & 31;
            *(v8bf*)&As[r][c] =
                *(const v8bf*)&ybuf[(size_t)(m0 + r) * CC + k0 + c];
        }
#pragma unroll
        for (int it = 0; it < 2; ++it) {
            int i = it * 1024 + tid * 4;
            int kk = i >> 6, nn = i & 63;
            float4 f = *(const float4*)&Wp[(size_t)(k0 + kk) * CC + n0 + nn];
            Bt[nn + 0][kk] = f2bf(f.x); Bt[nn + 1][kk] = f2bf(f.y);
            Bt[nn + 2][kk] = f2bf(f.z); Bt[nn + 3][kk] = f2bf(f.w);
        }
        __syncthreads();
        v16bf a = load_a_frag(&As[wave * 16][0], 32, lane);
        v16bf bfr[4];
#pragma unroll
        for (int s = 0; s < 4; ++s) bfr[s] = load_b_frag(&Bt[s * 16][0], 40, lane);
#pragma unroll
        for (int s = 0; s < 4; ++s) acc[s] = wmma_bf16(a, bfr[s], acc[s]);
        __syncthreads();
    }

#pragma unroll
    for (int s = 0; s < 4; ++s)
#pragma unroll
        for (int r = 0; r < 8; ++r) {
            int t = m0 + wave * 16 + grp * 8 + r;
            out[(size_t)t * CC + n0 + s * 16 + nl] = acc[s][r];
        }
}

// ---------------------------------------------------------------------------
extern "C" void kernel_launch(void* const* d_in, const int* in_sizes, int n_in,
                              void* d_out, int out_size, void* d_ws, size_t ws_size,
                              hipStream_t stream)
{
    (void)in_sizes; (void)n_in; (void)out_size; (void)ws_size;
    const float* x        = (const float*)d_in[0];
    const float* cosT     = (const float*)d_in[1];
    const float* sinT     = (const float*)d_in[2];
    const float* Wqkv     = (const float*)d_in[3];
    const float* Wproj    = (const float*)d_in[4];
    const float* dt_logit = (const float*)d_in[5];
    const float* kappa_un = (const float*)d_in[6];
    const float* xi_un    = (const float*)d_in[7];

    char* ws = (char*)d_ws;
    const size_t szHTD_bf = (size_t)HN * TT * DH * sizeof(bf16_t);
    const size_t szHTD_f  = (size_t)HN * TT * DH * sizeof(float);
    const size_t szHT_f   = (size_t)HN * TT * sizeof(float);
    bf16_t* qh   = (bf16_t*)(ws);
    bf16_t* kh   = (bf16_t*)(ws + szHTD_bf);
    bf16_t* vh   = (bf16_t*)(ws + 2 * szHTD_bf);
    float*  pv1  = (float*)(ws + 3 * szHTD_bf);
    float*  msb  = (float*)(ws + 3 * szHTD_bf + szHTD_f);
    float*  lsb  = (float*)(ws + 3 * szHTD_bf + szHTD_f + szHT_f);
    bf16_t* ybuf = (bf16_t*)(ws + 3 * szHTD_bf + szHTD_f + 2 * szHT_f);
    float*  out  = (float*)d_out;

    qkv_rope_kernel<<<dim3(TT / 128, 36), 256, 0, stream>>>(
        x, Wqkv, cosT, sinT, qh, kh, vh);
    attn_pass1_kernel<<<dim3(TT / 128, HN), 256, 0, stream>>>(
        qh, kh, vh, pv1, msb, lsb, kappa_un, xi_un);
    attn_pass2_kernel<<<dim3(TT / 128, HN), 256, 0, stream>>>(
        qh, kh, vh, pv1, msb, lsb, kappa_un, xi_un, dt_logit, ybuf);
    proj_kernel<<<dim3(TT / 128, CC / 64), 256, 0, stream>>>(
        ybuf, Wproj, out);
}
